// MvPDecoderLayer_66099546685937
// MI455X (gfx1250) — compile-verified, matrix operand order checked by reading, not run on hardware
//
#include <hip/hip_runtime.h>

// ---------------- problem constants ----------------
static constexpr int cNV  = 8;      // views (batch)
static constexpr int cLQ  = 420;    // queries
static constexpr int cNH  = 8;      // heads
static constexpr int cDFF = 1024;   // ffn dim
static constexpr int cP   = 21760;  // total pixels per view

typedef __attribute__((ext_vector_type(16))) _Float16 v16h;
typedef __attribute__((ext_vector_type(4)))  _Float16 v4h;
typedef __attribute__((ext_vector_type(8)))  float    v8f;

#define AS_LD 40   // padded LDS row stride for A tiles (halves)
#define BS_LD 40   // padded LDS column stride for B tiles (halves)

// ---------------- WMMA core helpers ----------------
__device__ __forceinline__ void wmma_step(const _Float16* As, const _Float16* Bs,
                                          int wm, int wn, int lane, v8f (&acc)[2][2]) {
  const int m  = lane & 15;
  const int kh = (lane >> 4) << 3;
  v16h a[2], b[2];
#pragma unroll
  for (int t = 0; t < 2; ++t) {
    const _Float16* ap = As + (wm + t * 16 + m) * AS_LD;
    const _Float16* bp = Bs + (wn + t * 16 + m) * BS_LD;
#pragma unroll
    for (int e = 0; e < 8; ++e) {
      a[t][e]     = ap[kh + e];
      a[t][8 + e] = ap[16 + kh + e];
      b[t][e]     = bp[kh + e];
      b[t][8 + e] = bp[16 + kh + e];
    }
  }
#pragma unroll
  for (int i = 0; i < 2; ++i)
#pragma unroll
    for (int j = 0; j < 2; ++j)
      acc[i][j] = __builtin_amdgcn_wmma_f32_16x16x32_f16(
          false, a[i], false, b[j], (short)0, acc[i][j], false, false);
}

__device__ __forceinline__ void store_tiles(float* C, int ldc, int bM, int bN,
                                            int wm, int wn, int lane,
                                            const v8f (&acc)[2][2], int M, int N,
                                            const float* bias, float alpha, int relu) {
  const int cl = lane & 15;
  const int rh = (lane >> 4) << 3;
#pragma unroll
  for (int i = 0; i < 2; ++i)
#pragma unroll
    for (int j = 0; j < 2; ++j)
#pragma unroll
      for (int v = 0; v < 8; ++v) {
        int gr = bM + wm + i * 16 + rh + v;
        int gc = bN + wn + j * 16 + cl;
        if (gr < M && gc < N) {
          float val = acc[i][j][v] * alpha;
          if (bias) val += bias[gc];
          if (relu) val = fmaxf(val, 0.f);
          C[(long long)gr * ldc + gc] = val;
        }
      }
}

// ---------------- specialized strided batched WMMA GEMM ----------------
// All operands are overread-safe (workspace-backed / padded): tile loads are
// unconditional float4; only KTAIL instantiations zero-fill the ragged K step.
struct GemmArgs {
  const float* A; const float* B; const float* bias; float* C;
  int M, N, K;
  int lda;                 // A row stride (k contiguous)
  long long ldb;           // BKIND=0: B row stride (n contig); BKIND=1: B col stride (k contig)
  int ldc;
  long long An, Ah, Bn, Bh, Cn, Ch;
  int nh;
  float alpha;
  int relu;
};

template <int BKIND, bool KTAIL>
__global__ void gemm_tiled(GemmArgs g) {
  __shared__ _Float16 As[64 * AS_LD];
  __shared__ _Float16 Bs[64 * BS_LD];
  const int tid  = threadIdx.x;
  const int wave = tid >> 5, lane = tid & 31;
  const int bM = blockIdx.x * 64, bN = blockIdx.y * 64;
  const int z  = blockIdx.z;
  const int zn = z / g.nh, zh = z % g.nh;
  const float* A = g.A + (long long)zn * g.An + (long long)zh * g.Ah;
  const float* B = g.B + (long long)zn * g.Bn + (long long)zh * g.Bh;
  float*       C = g.C + (long long)zn * g.Cn + (long long)zh * g.Ch;
  const int wm = (wave >> 1) * 32, wn = (wave & 1) * 32;
  v8f acc[2][2] = {};
  for (int k0 = 0; k0 < g.K; k0 += 32) {
    // ---- A tile 64x32 ----
    if (!KTAIL || (k0 + 32 <= g.K)) {
#pragma unroll
      for (int i = 0; i < 4; ++i) {
        int idx4 = tid + i * 128;
        int r = idx4 >> 3, kk = (idx4 & 7) * 4;
        const float* ap = A + (long long)(bM + r) * g.lda + k0 + kk;
        float4 v = *reinterpret_cast<const float4*>(ap);
        if (i == 0) __builtin_prefetch(ap + 32, 0, 1);
        v4h h4 = {(_Float16)v.x, (_Float16)v.y, (_Float16)v.z, (_Float16)v.w};
        *reinterpret_cast<v4h*>(&As[r * AS_LD + kk]) = h4;
      }
    } else {
#pragma unroll
      for (int i = 0; i < 16; ++i) {    // ragged K tail: zero-fill (rows overread OK)
        int idx = tid + i * 128;
        int r = idx >> 5, kk = idx & 31;
        int gk = k0 + kk;
        int gkc = gk < g.K ? gk : g.K - 1;
        float v = A[(long long)(bM + r) * g.lda + gkc];
        As[r * AS_LD + kk] = (_Float16)(gk < g.K ? v : 0.f);
      }
    }
    // ---- B tile 32x64 (column-major LDS) ----
    if (BKIND == 0) {                    // row-major B: float4 along columns
#pragma unroll
      for (int i = 0; i < 4; ++i) {
        int idx4 = tid + i * 128;
        int kk = idx4 >> 4, c = (idx4 & 15) * 4;
        const float* bp = B + (long long)(k0 + kk) * g.ldb + bN + c;
        float4 v = *reinterpret_cast<const float4*>(bp);
        if (i == 0) __builtin_prefetch(bp + 32 * g.ldb, 0, 1);
        Bs[(c + 0) * BS_LD + kk] = (_Float16)v.x;
        Bs[(c + 1) * BS_LD + kk] = (_Float16)v.y;
        Bs[(c + 2) * BS_LD + kk] = (_Float16)v.z;
        Bs[(c + 3) * BS_LD + kk] = (_Float16)v.w;
      }
    } else {                             // k-contiguous B: float4 along k
#pragma unroll
      for (int i = 0; i < 4; ++i) {
        int idx4 = tid + i * 128;
        int c = idx4 >> 3, kk = (idx4 & 7) * 4;
        float4 v = *reinterpret_cast<const float4*>(B + (long long)(bN + c) * g.ldb + k0 + kk);
        v4h h4 = {(_Float16)v.x, (_Float16)v.y, (_Float16)v.z, (_Float16)v.w};
        *reinterpret_cast<v4h*>(&Bs[c * BS_LD + kk]) = h4;
      }
    }
    __syncthreads();
    wmma_step(As, Bs, wm, wn, lane, acc);
    __syncthreads();
  }
  store_tiles(C, g.ldc, bM, bN, wm, wn, lane, acc, g.M, g.N, g.bias, g.alpha, g.relu);
}

// ---------------- rayconv value GEMM (fused channel-major gather A) ----------------
struct VGArgs {
  const float* src0; const float* src1; const float* src2; const float* src3;
  const float* cam0; const float* cam1; const float* cam2; const float* cam3;
  const float* W; const float* bias; float* C;
};

__global__ void value_gemm_wmma(VGArgs g) {
  __shared__ _Float16 As[64 * AS_LD];
  __shared__ _Float16 Bs[64 * BS_LD];
  const int tid  = threadIdx.x;
  const int wave = tid >> 5, lane = tid & 31;
  const int bM = blockIdx.x * 64, bN = blockIdx.y * 64;
  const int n  = blockIdx.z;
  const int wm = (wave >> 1) * 32, wn = (wave & 1) * 32;
  const float* sp; const float* cp; int HW; int startl;
  if (bM >= 21504)      { sp = g.src3; cp = g.cam3; HW = 256;   startl = 21504; }
  else if (bM >= 20480) { sp = g.src2; cp = g.cam2; HW = 1024;  startl = 20480; }
  else if (bM >= 16384) { sp = g.src1; cp = g.cam1; HW = 4096;  startl = 16384; }
  else                  { sp = g.src0; cp = g.cam0; HW = 16384; startl = 0; }
  const int pl0 = bM - startl;
  const float* srow = sp + (long long)n * 256 * HW + pl0;      // + k*HW + r
  const float* crow = cp + ((long long)n * HW + pl0) * 3;      // + r*3 + kc
  v8f acc[2][2] = {};
  for (int k0 = 0; k0 < 259; k0 += 32) {
    if (k0 < 256) {
#pragma unroll
      for (int i = 0; i < 4; ++i) {      // A: float4 along p, unconditional
        int idx4 = tid + i * 128;
        int kk = idx4 >> 4, r = (idx4 & 15) * 4;
        const float* ap = srow + (long long)(k0 + kk) * HW + r;
        float4 v = *reinterpret_cast<const float4*>(ap);
        if (i == 0) __builtin_prefetch(ap + 32LL * HW, 0, 1);
        As[(r + 0) * AS_LD + kk] = (_Float16)v.x;
        As[(r + 1) * AS_LD + kk] = (_Float16)v.y;
        As[(r + 2) * AS_LD + kk] = (_Float16)v.z;
        As[(r + 3) * AS_LD + kk] = (_Float16)v.w;
      }
#pragma unroll
      for (int i = 0; i < 4; ++i) {      // B: rayconv_w row-major
        int idx4 = tid + i * 128;
        int kk = idx4 >> 4, c = (idx4 & 15) * 4;
        float4 v = *reinterpret_cast<const float4*>(g.W + (long long)(k0 + kk) * 256 + bN + c);
        Bs[(c + 0) * BS_LD + kk] = (_Float16)v.x;
        Bs[(c + 1) * BS_LD + kk] = (_Float16)v.y;
        Bs[(c + 2) * BS_LD + kk] = (_Float16)v.z;
        Bs[(c + 3) * BS_LD + kk] = (_Float16)v.w;
      }
    } else {
#pragma unroll
      for (int i = 0; i < 16; ++i) {     // cam tail: k = 256..258
        int idx = tid + i * 128;
        int r = idx & 63, kk = idx >> 6;
        int kc = kk < 3 ? kk : 2;
        float v = crow[r * 3 + kc];
        As[r * AS_LD + kk] = (_Float16)(kk < 3 ? v : 0.f);
      }
#pragma unroll
      for (int i = 0; i < 16; ++i) {
        int idx = tid + i * 128;
        int c = idx & 63, kk = idx >> 6;
        int gk = k0 + kk;
        int gkc = gk < 259 ? gk : 258;
        float v = g.W[(long long)gkc * 256 + bN + c];
        Bs[c * BS_LD + kk] = (_Float16)(gk < 259 ? v : 0.f);
      }
    }
    __syncthreads();
    wmma_step(As, Bs, wm, wn, lane, acc);
    __syncthreads();
  }
  float* C = g.C + (long long)n * cP * 256;
  store_tiles(C, 256, bM, bN, wm, wn, lane, acc, cP, 256, g.bias, 1.0f, 0);
}

// ---------------- elementwise / reduction kernels ----------------
__global__ void add2_kernel(const float* a, const float* b, float* o, int n) {
  int i = blockIdx.x * blockDim.x + threadIdx.x;
  if (i < n) o[i] = a[i] + b[i];
}

__global__ void copy_kernel(const float* a, float* o, int n) {
  int i = blockIdx.x * blockDim.x + threadIdx.x;
  if (i < n) o[i] = a[i];
}

// zero-pad aw_w (256x32) into AWP (256x64)
__global__ void pad_aw_kernel(const float* w, float* o) {
  int i = blockIdx.x * blockDim.x + threadIdx.x;   // 256*64
  int k = i >> 6, c = i & 63;
  o[i] = (c < 32) ? w[k * 32 + c] : 0.f;
}

__global__ void layernorm_add(const float* a, const float* b, const float* g,
                              const float* be, float* out) {
  __shared__ float red[256];
  long long row = blockIdx.x;
  int t = threadIdx.x;
  float x = a[row * 256 + t] + b[row * 256 + t];
  red[t] = x; __syncthreads();
  for (int s = 128; s > 0; s >>= 1) { if (t < s) red[t] += red[t + s]; __syncthreads(); }
  float m = red[0] * (1.f / 256.f); __syncthreads();
  float d = x - m;
  red[t] = d * d; __syncthreads();
  for (int s = 128; s > 0; s >>= 1) { if (t < s) red[t] += red[t + s]; __syncthreads(); }
  float var = red[0] * (1.f / 256.f);
  out[row * 256 + t] = d * rsqrtf(var + 1e-5f) * g[t] + be[t];
}

__global__ void softmax_row420(float* buf) {
  __shared__ float red[256];
  long long row = blockIdx.x;
  float* p = buf + row * cLQ;
  int t = threadIdx.x;
  float a  = (t < cLQ)       ? p[t]       : -3.0e38f;
  float b2 = (t + 256 < cLQ) ? p[t + 256] : -3.0e38f;
  red[t] = fmaxf(a, b2); __syncthreads();
  for (int s = 128; s > 0; s >>= 1) { if (t < s) red[t] = fmaxf(red[t], red[t + s]); __syncthreads(); }
  float mx = red[0]; __syncthreads();
  float ea = (t < cLQ)       ? __expf(a  - mx) : 0.f;
  float eb = (t + 256 < cLQ) ? __expf(b2 - mx) : 0.f;
  red[t] = ea + eb; __syncthreads();
  for (int s = 128; s > 0; s >>= 1) { if (t < s) red[t] += red[t + s]; __syncthreads(); }
  float inv = 1.f / red[0];
  if (t < cLQ)       p[t]       = ea * inv;
  if (t + 256 < cLQ) p[t + 256] = eb * inv;
}

struct RefArgs { const float* s0; const float* s1; const float* s2; const float* s3;
                 const float* ref; const float* q2; float* qf; };
__global__ void ref_qf_kernel(RefArgs g) {
  int b  = blockIdx.x;          // (n*LQ+q)*4 + l
  int l  = b & 3;
  int nq = b >> 2;
  int n  = nq / cLQ;
  int c  = threadIdx.x;
  const float* s;
  if (l == 0) s = g.s0; else if (l == 1) s = g.s1; else if (l == 2) s = g.s2; else s = g.s3;
  int W = 128 >> l;
  float gx = g.ref[((long long)nq * 4 + l) * 2 + 0] * 2.f - 1.f;
  float gy = g.ref[((long long)nq * 4 + l) * 2 + 1] * 2.f - 1.f;
  gx = fminf(fmaxf(gx, -1.1f), 1.1f);
  gy = fminf(fmaxf(gy, -1.1f), 1.1f);
  float x = ((gx + 1.f) * W - 1.f) * 0.5f;
  float y = ((gy + 1.f) * W - 1.f) * 0.5f;
  float x0f = floorf(x), y0f = floorf(y);
  int x0 = (int)x0f, y0 = (int)y0f;
  float wx = x - x0f, wy = y - y0f;
  const float* img = s + ((long long)n * 256 + c) * (W * W);
  auto fetch = [&](int xx, int yy) -> float {
    int xc = min(max(xx, 0), W - 1), yc = min(max(yy, 0), W - 1);
    float v = img[(long long)yc * W + xc];
    return (xx == xc && yy == yc) ? v : 0.f;
  };
  float v00 = fetch(x0, y0),     v01 = fetch(x0 + 1, y0);
  float v10 = fetch(x0, y0 + 1), v11 = fetch(x0 + 1, y0 + 1);
  float rf = (v00 * (1.f - wx) + v01 * wx) * (1.f - wy) +
             (v10 * (1.f - wx) + v11 * wx) * wy;
  g.qf[(long long)b * 256 + c] = rf + g.q2[(long long)nq * 256 + c];
}

__global__ void aw_softmax_kernel(const float* awr, float* awn) {
  int idx = blockIdx.x * 256 + threadIdx.x;
  if (idx >= cNV * cLQ * cNH) return;
  int h = idx & 7, nq = idx >> 3;
  const float* base = awr + ((long long)nq * 4 + (h >> 1)) * 32 + (h & 1) * 16;
  float mx = -3.0e38f;
#pragma unroll
  for (int k = 0; k < 16; ++k) mx = fmaxf(mx, base[k]);
  float e[16], s = 0.f;
#pragma unroll
  for (int k = 0; k < 16; ++k) { e[k] = __expf(base[k] - mx); s += e[k]; }
  float inv = 1.f / s;
  float* o = awn + ((long long)nq * 8 + h) * 16;
#pragma unroll
  for (int k = 0; k < 16; ++k) o[k] = e[k] * inv;
}

__global__ void sample_accum_kernel(const float* ref, const float* offb,
                                    const float* awn, const float* value, float* out) {
  int nq = blockIdx.x;
  int n  = nq / cLQ;
  int t  = threadIdx.x;
  int h  = t >> 5, d = t & 31;
  float acc = 0.f;
  long long offrow = (long long)nq * 4 + (h >> 1);
  const int START[4] = {0, 16384, 20480, 21504};
#pragma unroll
  for (int l = 0; l < 4; ++l) {
    int W = 128 >> l;
    float rx = ref[((long long)nq * 4 + l) * 2 + 0];
    float ry = ref[((long long)nq * 4 + l) * 2 + 1];
    long long vbase = ((long long)n * cP + START[l]) * 256 + h * 32 + d;
#pragma unroll
    for (int p = 0; p < 4; ++p) {
      int j = (h & 1) * 32 + l * 8 + p * 2;
      float ox = offb[offrow * 64 + j];
      float oy = offb[offrow * 64 + j + 1];
      float x = (rx + ox / (float)W) * W - 0.5f;
      float y = (ry + oy / (float)W) * W - 0.5f;
      float w = awn[((long long)nq * 8 + h) * 16 + l * 4 + p];
      float x0f = floorf(x), y0f = floorf(y);
      int x0 = (int)x0f, y0 = (int)y0f;
      float wx = x - x0f, wy = y - y0f;
      auto fetch = [&](int xx, int yy) -> float {
        int xc = min(max(xx, 0), W - 1), yc = min(max(yy, 0), W - 1);
        float v = value[vbase + ((long long)yc * W + xc) * 256];
        return (xx == xc && yy == yc) ? v : 0.f;
      };
      float v00 = fetch(x0, y0),     v01 = fetch(x0 + 1, y0);
      float v10 = fetch(x0, y0 + 1), v11 = fetch(x0 + 1, y0 + 1);
      float s = (v00 * (1.f - wx) + v01 * wx) * (1.f - wy) +
                (v10 * (1.f - wx) + v11 * wx) * wy;
      acc += w * s;
    }
  }
  out[(long long)nq * 256 + t] = acc;
}

// ---------------- host orchestration ----------------
template <int BKIND, bool KTAIL>
static inline void launch_gemm(hipStream_t s,
    const float* A, long long An, long long Ah, int lda,
    const float* B, long long Bn, long long Bh, long long ldb,
    float* C, long long Cn, long long Ch, int ldc,
    const float* bias, int M, int N, int K, int nb, int nh, float alpha, int relu) {
  GemmArgs g;
  g.A = A; g.B = B; g.bias = bias; g.C = C;
  g.M = M; g.N = N; g.K = K; g.lda = lda; g.ldb = ldb; g.ldc = ldc;
  g.An = An; g.Ah = Ah; g.Bn = Bn; g.Bh = Bh; g.Cn = Cn; g.Ch = Ch;
  g.nh = nh; g.alpha = alpha; g.relu = relu;
  dim3 grid((M + 63) / 64, (N + 63) / 64, nb * nh);
  gemm_tiled<BKIND, KTAIL><<<grid, dim3(128), 0, s>>>(g);
}

extern "C" void kernel_launch(void* const* d_in, const int* in_sizes, int n_in,
                              void* d_out, int out_size, void* d_ws, size_t ws_size,
                              hipStream_t stream) {
  (void)in_sizes; (void)n_in; (void)out_size; (void)ws_size;
  const float* query     = (const float*)d_in[0];
  const float* query_pos = (const float*)d_in[1];
  const float* refpts    = (const float*)d_in[2];
  const float* src[4] = {(const float*)d_in[3], (const float*)d_in[5],
                         (const float*)d_in[7], (const float*)d_in[9]};
  const float* cam[4] = {(const float*)d_in[4], (const float*)d_in[6],
                         (const float*)d_in[8], (const float*)d_in[10]};
  const float* rayconv_w = (const float*)d_in[13];
  const float* rayconv_b = (const float*)d_in[14];
  const float* off_w     = (const float*)d_in[15];
  const float* off_b     = (const float*)d_in[16];
  const float* aw_w      = (const float*)d_in[17];
  const float* aw_b      = (const float*)d_in[18];
  const float* outp_w    = (const float*)d_in[19];
  const float* outp_b    = (const float*)d_in[20];
  const float* sa_wq = (const float*)d_in[21]; const float* sa_bq = (const float*)d_in[22];
  const float* sa_wk = (const float*)d_in[23]; const float* sa_bk = (const float*)d_in[24];
  const float* sa_wv = (const float*)d_in[25]; const float* sa_bv = (const float*)d_in[26];
  const float* sa_wo = (const float*)d_in[27]; const float* sa_bo = (const float*)d_in[28];
  const float* ln1_g = (const float*)d_in[29]; const float* ln1_b = (const float*)d_in[30];
  const float* ln2_g = (const float*)d_in[31]; const float* ln2_b = (const float*)d_in[32];
  const float* ln3_g = (const float*)d_in[33]; const float* ln3_b = (const float*)d_in[34];
  const float* ff1_w = (const float*)d_in[35]; const float* ff1_b = (const float*)d_in[36];
  const float* ff2_w = (const float*)d_in[37]; const float* ff2_b = (const float*)d_in[38];

  const int M1 = cNV * cLQ;                 // 3360
  float* ws = (float*)d_ws;
  size_t cur = 0;
  auto alloc = [&](size_t nfl) { float* p = ws + cur; cur += nfl; return p; };
  float* QK  = alloc((size_t)M1 * 256);
  float* QC  = alloc((size_t)M1 * 256);     // ws copy of query (overread-safe A)
  float* Qb  = alloc((size_t)M1 * 256);
  float* Kb  = alloc((size_t)M1 * 256);
  float* Vb  = alloc((size_t)M1 * 256);
  float* SC  = alloc((size_t)cNV * cNH * cLQ * cLQ);
  float* AV  = alloc((size_t)M1 * 256);
  float* SA  = alloc((size_t)M1 * 256);
  float* TG  = alloc((size_t)M1 * 256);
  float* Q2  = alloc((size_t)M1 * 256);
  float* QF  = alloc((size_t)M1 * 4 * 256);
  float* OF  = alloc((size_t)M1 * 4 * 64);
  float* AWP = alloc((size_t)256 * 64);     // zero-padded aw_w
  float* AWR = alloc((size_t)M1 * 4 * 32);
  float* AWN = alloc((size_t)M1 * cNH * 16);
  float* VAL = alloc((size_t)cNV * cP * 256);
  float* SP  = alloc((size_t)M1 * 256);
  float* PA  = alloc((size_t)M1 * 256);
  float* T2  = alloc((size_t)M1 * 256);
  float* FH  = alloc((size_t)M1 * cDFF);
  float* FO  = alloc((size_t)M1 * 256);
  (void)alloc(32768);  // overread guard slack

  const int nElem = M1 * 256;
  const float scale = 0.17677669529663687f;  // 1/sqrt(32)

  // ---- self attention ----
  add2_kernel<<<(nElem + 255) / 256, 256, 0, stream>>>(query, query_pos, QK, nElem);
  copy_kernel<<<(nElem + 255) / 256, 256, 0, stream>>>(query, QC, nElem);
  launch_gemm<0, false>(stream, QK, 0, 0, 256, sa_wq, 0, 0, 256,
                        Qb, 0, 0, 256, sa_bq, M1, 256, 256, 1, 1, 1.f, 0);
  launch_gemm<0, false>(stream, QK, 0, 0, 256, sa_wk, 0, 0, 256,
                        Kb, 0, 0, 256, sa_bk, M1, 256, 256, 1, 1, 1.f, 0);
  launch_gemm<0, false>(stream, QC, 0, 0, 256, sa_wv, 0, 0, 256,
                        Vb, 0, 0, 256, sa_bv, M1, 256, 256, 1, 1, 1.f, 0);
  // scores[n,h,q,k] = scale * Q . K  (B = K^T : k-contiguous)
  launch_gemm<1, false>(stream, Qb, (long long)cLQ * 256, 32, 256,
                        Kb, (long long)cLQ * 256, 32, /*ldb=*/256,
                        SC, (long long)cNH * cLQ * cLQ, (long long)cLQ * cLQ, cLQ,
                        nullptr, cLQ, cLQ, 32, cNV, cNH, scale, 0);
  softmax_row420<<<cNV * cNH * cLQ, 256, 0, stream>>>(SC);
  // AV[n,q,h,d] = att @ V  (K = 420, ragged)
  launch_gemm<0, true>(stream, SC, (long long)cNH * cLQ * cLQ, (long long)cLQ * cLQ, cLQ,
                       Vb, (long long)cLQ * 256, 32, /*ldb=*/256,
                       AV, (long long)cLQ * 256, 32, 256,
                       nullptr, cLQ, 32, cLQ, cNV, cNH, 1.f, 0);
  launch_gemm<0, false>(stream, AV, 0, 0, 256, sa_wo, 0, 0, 256,
                        SA, 0, 0, 256, sa_bo, M1, 256, 256, 1, 1, 1.f, 0);
  layernorm_add<<<M1, 256, 0, stream>>>(query, SA, ln2_g, ln2_b, TG);

  // ---- projective attention ----
  add2_kernel<<<(nElem + 255) / 256, 256, 0, stream>>>(TG, query_pos, Q2, nElem);
  RefArgs ra{src[0], src[1], src[2], src[3], refpts, Q2, QF};
  ref_qf_kernel<<<M1 * 4, 256, 0, stream>>>(ra);
  launch_gemm<0, false>(stream, QF, 0, 0, 256, off_w, 0, 0, 64,
                        OF, 0, 0, 64, off_b, M1 * 4, 64, 256, 1, 1, 1.f, 0);
  pad_aw_kernel<<<(256 * 64 + 255) / 256, 256, 0, stream>>>(aw_w, AWP);
  launch_gemm<0, false>(stream, QF, 0, 0, 256, AWP, 0, 0, 64,
                        AWR, 0, 0, 32, aw_b, M1 * 4, 32, 256, 1, 1, 1.f, 0);
  aw_softmax_kernel<<<(cNV * cLQ * cNH + 255) / 256, 256, 0, stream>>>(AWR, AWN);
  VGArgs vg{src[0], src[1], src[2], src[3], cam[0], cam[1], cam[2], cam[3],
            rayconv_w, rayconv_b, VAL};
  value_gemm_wmma<<<dim3(cP / 64, 4, cNV), dim3(128), 0, stream>>>(vg);
  sample_accum_kernel<<<M1, 256, 0, stream>>>(refpts, OF, AWN, VAL, SP);
  launch_gemm<0, false>(stream, SP, 0, 0, 256, outp_w, 0, 0, 256,
                        PA, 0, 0, 256, outp_b, M1, 256, 256, 1, 1, 1.f, 0);
  layernorm_add<<<M1, 256, 0, stream>>>(TG, PA, ln1_g, ln1_b, T2);

  // ---- FFN ----
  launch_gemm<0, false>(stream, T2, 0, 0, 256, ff1_w, 0, 0, cDFF,
                        FH, 0, 0, cDFF, ff1_b, M1, cDFF, 256, 1, 1, 1.f, 1);
  launch_gemm<0, false>(stream, FH, 0, 0, cDFF, ff2_w, 0, 0, 256,
                        FO, 0, 0, 256, ff2_b, M1, 256, cDFF, 1, 1, 1.f, 0);
  layernorm_add<<<M1, 256, 0, stream>>>(T2, FO, ln3_g, ln3_b, (float*)d_out);
}
